// MemoryEfficientPatchDownScale_14516989460759
// MI455X (gfx1250) — matrix operand order repaired and checked by reading, not compiled, not run on hardware
//
#include <hip/hip_runtime.h>

// bf16 WMMA fragment types (CDNA5 wave32)
typedef __attribute__((ext_vector_type(16))) __bf16 v16bf;
typedef __attribute__((ext_vector_type(8)))  __bf16 v8bf;
typedef __attribute__((ext_vector_type(8)))  float  v8f;

#define IN_C   64
#define OUT_C  128
#define HIDDEN 512
#define FAN_IN 256   // IN_C * 2 * 2
#define Hdim   256
#define Wdim   256
#define H2d    128
#define W2d    128
#define Bsz    16

// Fused: patchify(2x2) -> Linear(256->512)+SiLU -> Linear(512->128) -> +2x2-mean residual
// One wave = 16 patch rows. One workgroup (8 waves) = 128 rows. Grid = M/128.
__global__ __launch_bounds__(256)
void fused_patch_mlp(const float* __restrict__ x,
                     const float* __restrict__ w1,
                     const float* __restrict__ b1,
                     const float* __restrict__ w2,
                     const float* __restrict__ b2,
                     float* __restrict__ out)
{
    __shared__ __bf16 w1s[16 * FAN_IN];      // 8 KB : 16 hidden rows x 256 K (bf16)
    __shared__ __bf16 w2s[OUT_C * 32];       // 8 KB : 128 out rows x 32 K2 (bf16)
    __shared__ __bf16 trs[8 * 256];          // 4 KB : per-wave 16x16 D->A transpose tile
    __shared__ float  pooled[128 * IN_C];    // 32 KB: per-row 2x2 mean, 64 channels

    const int tid  = threadIdx.x;
    const int wave = tid >> 5;
    const int lane = tid & 31;
    const int l16  = lane & 15;
    const int hi   = lane >> 4;              // which 16-lane half

    // 16 consecutive flattened patch rows per wave (W2d=128 keeps them in one (b,h2) line)
    const int mg0 = blockIdx.x * 128 + wave * 16;
    const int b   = mg0 >> 14;               // / (H2d*W2d)
    const int rem = mg0 & 16383;
    const int h2  = rem >> 7;
    const int w2b = rem & 127;

    // ---- Load patch matrix (16x256) straight into WMMA A-fragments (bf16) ----
    // A 16x32 bf16 layout: lane (M=l16), elems 0..7 -> K = 8*hi + e,
    //                                   elems 8..15 -> K = 16 + 8*hi + (e-8), per 32-K step.
    // K = c*4 + dy*2 + dx  =>  this lane covers channels with (c&3) in {2*hi, 2*hi+1}.
    v16bf a1[8];
    {
        const int px = 2 * (w2b + l16);      // this lane's patch column origin (dx=0)
        #pragma unroll
        for (int q = 0; q < 16; ++q) {
            const int ks = q >> 1;
            const int eb = (q & 1) * 8;
            #pragma unroll
            for (int t = 0; t < 2; ++t) {
                const int c = 4 * q + 2 * hi + t;
                float ps = 0.f;
                #pragma unroll
                for (int dy = 0; dy < 2; ++dy) {
                    const int idx = ((b * IN_C + c) * Hdim + (2 * h2 + dy)) * Wdim + px;
                    const float2 v = *reinterpret_cast<const float2*>(x + idx);
                    a1[ks][eb + 4 * t + 2 * dy + 0] = (__bf16)v.x;
                    a1[ks][eb + 4 * t + 2 * dy + 1] = (__bf16)v.y;
                    ps += v.x + v.y;
                }
                pooled[(wave * 16 + l16) * IN_C + c] = 0.25f * ps;   // 2x2 mean residual
            }
        }
    }

    // ---- Output accumulators: wave's 16 rows x 128 channels, bias2 pre-loaded ----
    v8f acc2[8];
    #pragma unroll
    for (int oc = 0; oc < 8; ++oc) {
        const float bv = b2[16 * oc + l16];
        #pragma unroll
        for (int r = 0; r < 8; ++r) acc2[oc][r] = bv;
    }

    // ---- Hidden loop: 16 chunks x 32 hidden values ----
    for (int hc = 0; hc < 16; ++hc) {
        v16bf a2;                            // 16x32 bf16 activation fragment for GEMM2
        #pragma unroll
        for (int sub = 0; sub < 2; ++sub) {
            const int h0 = hc * 32 + sub * 16;
            __syncthreads();
            // Cooperative stage of W1 rows [h0, h0+16) x 256 K, f32 -> bf16
            for (int i = tid; i < 16 * FAN_IN; i += 256)
                w1s[i] = (__bf16)w1[(h0 + (i >> 8)) * FAN_IN + (i & 255)];
            __syncthreads();
            // Prefetch the next W1 block into L2 while the matrix pipe works
            if ((h0 + 16) < HIDDEN)
                __builtin_prefetch(&w1[(h0 + 16) * FAN_IN + tid * 16], 0, 1);

            // GEMM1: acc1 (16x16, f32) = patch(16x256) x W1^T chunk (256x16) + bias1
            const float bv = b1[h0 + l16];
            v8f acc1;
            #pragma unroll
            for (int r = 0; r < 8; ++r) acc1[r] = bv;
            #pragma unroll
            for (int ks = 0; ks < 8; ++ks) {
                // B 32x16 bf16 layout: lane (N=l16), K = 8*hi + 0..7 and 16 + 8*hi + 0..7 within the step
                const __bf16* bp = w1s + l16 * FAN_IN + 32 * ks + 8 * hi;
                const v8bf blo = *reinterpret_cast<const v8bf*>(bp);
                const v8bf bh8 = *reinterpret_cast<const v8bf*>(bp + 16);
                v16bf bf;
                #pragma unroll
                for (int e = 0; e < 8; ++e) { bf[e] = blo[e]; bf[e + 8] = bh8[e]; }
                acc1 = __builtin_amdgcn_wmma_f32_16x16x32_bf16(
                    false, a1[ks], false, bf, (short)0, acc1, false, false);
            }
            // SiLU in-register (fast v_rcp_f32, matrix pipe only needs bf16 accuracy),
            // then D-layout -> A-layout relayout via per-wave LDS tile.
            // D layout: VGPR r holds (M = r + 8*hi, N = l16).
            #pragma unroll
            for (int r = 0; r < 8; ++r) {
                const float o = acc1[r];
                const float s = o * __builtin_amdgcn_rcpf(1.0f + __expf(-o));
                trs[wave * 256 + (r + 8 * hi) * 16 + l16] = (__bf16)s;
            }
            // A layout read-back: row M=l16, local K = 8*hi + 0..7 (intra-wave, DScnt-ordered)
            const v8bf t8 = *reinterpret_cast<const v8bf*>(trs + wave * 256 + l16 * 16 + 8 * hi);
            #pragma unroll
            for (int e = 0; e < 8; ++e) a2[8 * sub + e] = t8[e];
        }

        __syncthreads();
        // Cooperative stage of W2 chunk: all 128 out rows x K2 in [hc*32, +32), f32 -> bf16
        for (int i = tid; i < OUT_C * 32; i += 256)
            w2s[i] = (__bf16)w2[(i >> 5) * HIDDEN + hc * 32 + (i & 31)];
        __syncthreads();

        // GEMM2: accumulate all 8 output-channel tiles with this 32-hidden slab
        #pragma unroll
        for (int oc = 0; oc < 8; ++oc) {
            const __bf16* bp = w2s + (16 * oc + l16) * 32 + 8 * hi;
            const v8bf blo = *reinterpret_cast<const v8bf*>(bp);
            const v8bf bh8 = *reinterpret_cast<const v8bf*>(bp + 16);
            v16bf bf;
            #pragma unroll
            for (int e = 0; e < 8; ++e) { bf[e] = blo[e]; bf[e + 8] = bh8[e]; }
            acc2[oc] = __builtin_amdgcn_wmma_f32_16x16x32_bf16(
                false, a2, false, bf, (short)0, acc2[oc], false, false);
        }
    }

    // ---- Residual add + wide coalesced store ----
    // Lane holds 8 consecutive w2 positions (m = 8*hi + r) for one channel ch = 16*oc + l16,
    // so each oc-chunk stores as two 16B-aligned float4 (global_store_b128).
    #pragma unroll
    for (int oc = 0; oc < 8; ++oc) {
        const int ch = 16 * oc + l16;
        float v[8];
        #pragma unroll
        for (int r = 0; r < 8; ++r) {
            const int m = 8 * hi + r;
            v[r] = acc2[oc][r] + pooled[(wave * 16 + m) * IN_C + (ch >> 1)];
        }
        float* p = out + ((b * OUT_C + ch) * H2d + h2) * W2d + w2b + 8 * hi;
        const float4 f0 = {v[0], v[1], v[2], v[3]};
        const float4 f1 = {v[4], v[5], v[6], v[7]};
        *reinterpret_cast<float4*>(p)     = f0;
        *reinterpret_cast<float4*>(p + 4) = f1;
    }
}

extern "C" void kernel_launch(void* const* d_in, const int* in_sizes, int n_in,
                              void* d_out, int out_size, void* d_ws, size_t ws_size,
                              hipStream_t stream) {
    (void)in_sizes; (void)n_in; (void)out_size; (void)d_ws; (void)ws_size;
    const float* x  = (const float*)d_in[0];
    const float* w1 = (const float*)d_in[1];
    const float* b1 = (const float*)d_in[2];
    const float* w2 = (const float*)d_in[3];
    const float* b2 = (const float*)d_in[4];
    float* out = (float*)d_out;

    const int M = Bsz * H2d * W2d;           // 262144 patch rows
    dim3 grid(M / 128), block(256);
    hipLaunchKernelGGL(fused_patch_mlp, grid, block, 0, stream, x, w1, b1, w2, b2, out);
}
